// HMLET_End_52123723104419
// MI455X (gfx1250) — matrix operand order repaired and checked by reading, not compiled, not run on hardware
//
#include <hip/hip_runtime.h>
#include <hip/hip_bf16.h>

// ---------------- problem constants (match reference) ----------------
#define NUSERS 100000
#define NITEMS 50000
#define NTOT   150000           // N = NUSERS + NITEMS, divisible by 16
#define DIM    128
#define EMB_STRIDE (5 * DIM)    // embs is [N,5,128] row-major
#define NNZ_K  3000000
#define BSZ    16384
#define WAVES_PER_BLOCK 8

typedef __attribute__((ext_vector_type(16))) _Float16 v16h;
typedef __attribute__((ext_vector_type(8)))  _Float16 v8h;
typedef __attribute__((ext_vector_type(8)))  float    v8f;

union CAcc { v8f v; float f[8]; };

__device__ __forceinline__ float elu_f(float x) {
    return x > 0.f ? x : (__expf(x) - 1.f);
}

// ---- WMMA fragment helpers (CDNA5 16x16x32 f16 layouts, wave32) ----
// A (16x32 f16): lane row M=lane&15; halves 0..7 = K kh..kh+7, halves 8..15 = K kh+16..kh+23
// (kh = 8*(lane>=16)).  Both runs are contiguous -> two 16B loads + shuffle.
__device__ __forceinline__ v16h a_frag_lds(const _Float16* X, int ld, int lane, int kbase) {
    int r  = lane & 15;
    int kh = (lane >> 4) * 8;
    const _Float16* p = X + r * ld + kbase + kh;
    v8h lo = *(const v8h*)(p);        // ds_load_b128
    v8h hi = *(const v8h*)(p + 16);   // ds_load_b128
    return __builtin_shufflevector(lo, hi, 0, 1, 2, 3, 4, 5, 6, 7,
                                   8, 9, 10, 11, 12, 13, 14, 15);
}

// B fragments pre-swizzled to fragment-major: frag(c,t) = 32 lanes x 16 contiguous halves.
__device__ __forceinline__ v16h b_frag_swz(const _Float16* Wswz, int NT, int c, int t, int lane) {
    return *(const v16h*)(Wswz + (((c * NT + t) * 32 + lane) << 4));  // 2x global_load_b128
}

// C/D (16x16 f32, 8 VGPRs): lane 0-15 -> N=lane, M=r; lane 16-31 -> N=lane-16, M=r+8.
__device__ __forceinline__ void store_h_f16(_Float16* H, int ld, int lane, int nbase,
                                            CAcc acc, const float* bptr) {
    int n  = nbase + (lane & 15);
    int mh = (lane >> 4) * 8;
    float bc = bptr[n];
#pragma unroll
    for (int r = 0; r < 8; ++r) {
        float v = acc.f[r] + bc;
        v = v > 0.f ? v : 0.f;
        H[(mh + r) * ld + n] = (_Float16)v;
    }
}

__device__ __forceinline__ void store_h_f32(float* H, int ld, int lane, int nbase,
                                            CAcc acc, const float* bptr) {
    int n  = nbase + (lane & 15);
    int mh = (lane >> 4) * 8;
    float bc = bptr[n];
#pragma unroll
    for (int r = 0; r < 8; ++r) {
        float v = acc.f[r] + bc;
        v = v > 0.f ? v : 0.f;
        H[(mh + r) * ld + n] = v;
    }
}

// ---------------- weight preparation kernels ----------------
__global__ void k_cvt_f16(const float* __restrict__ src, _Float16* __restrict__ dst, int n) {
    int i = blockIdx.x * blockDim.x + threadIdx.x;
    if (i < n) dst[i] = (_Float16)src[i];
}

// Swizzle a KxN f32 weight into fragment-major f16:
// dst[((c*NT + t)*32 + lane)*16 + i] = W[(c*32 + (lane>>4)*8 + (i&7) + (i>>3)*16) * N + t*16 + (lane&15)]
__global__ void k_swz_f16(const float* __restrict__ src, _Float16* __restrict__ dst,
                          int N, int total) {
    int idx = blockIdx.x * blockDim.x + threadIdx.x;
    if (idx >= total) return;
    int i    = idx & 15;
    int lane = (idx >> 4) & 31;
    int ft   = idx >> 9;          // fragment index = c*NT + t
    int NT   = N >> 4;
    int c    = ft / NT;
    int t    = ft - c * NT;
    int k    = c * 32 + (lane >> 4) * 8 + (i & 7) + (i >> 3) * 16;
    int n    = t * 16 + (lane & 15);
    dst[idx] = (_Float16)src[k * N + n];
}

// Writes e0 into slot 0 of embs, zeroes SpMM accumulators (slots 1,2 + ws lin3/lin4)
// and the four count scalars. Must run every launch (harness poisons buffers).
__global__ void k_init(const float* __restrict__ uemb, const float* __restrict__ iemb,
                       float* __restrict__ embs, float* __restrict__ lin3,
                       float* __restrict__ lin4, float* __restrict__ cnts) {
    long i = (long)blockIdx.x * blockDim.x + threadIdx.x;
    const long total = (long)NTOT * DIM;
    if (i < total) {
        int n = (int)(i / DIM), d = (int)(i % DIM);
        float v = (n < NUSERS) ? uemb[(long)n * DIM + d]
                               : iemb[(long)(n - NUSERS) * DIM + d];
        float* row = embs + (long)n * EMB_STRIDE;
        row[d]           = v;    // slot 0: e0
        row[DIM + d]     = 0.f;  // slot 1: e_lin1 accumulator
        row[2 * DIM + d] = 0.f;  // slot 2: e_lin2 accumulator
        lin3[i] = 0.f;
        lin4[i] = 0.f;
    }
    if (i < 4) cnts[i] = 0.f;
}

// ---------------- SpMM: one wave per edge, 32 lanes x float4 = 128-wide row ----------------
// Gathers/scatters are L2-resident (76.8 MB table << 192 MB L2); HBM only streams edge lists.
__global__ void k_spmm(const float* __restrict__ srcA, const float* __restrict__ srcB,
                       int split, int strA, int strB,
                       const float* __restrict__ vals, const int* __restrict__ rows,
                       const int* __restrict__ cols,
                       float* __restrict__ dst, int strDst, int nnz) {
    int lane   = threadIdx.x & 31;
    int wave   = blockIdx.x * (blockDim.x >> 5) + (threadIdx.x >> 5);
    int nwaves = gridDim.x * (blockDim.x >> 5);
    for (int e = wave; e < nnz; e += nwaves) {
        int   c = cols[e];
        int   r = rows[e];
        float v = vals[e];
        const float* s = (c < split) ? (srcA + (long)c * strA)
                                     : (srcB + (long)(c - split) * strB);
        // prefetch next gather row (global_prefetch_b8)
        int e2 = e + nwaves;
        if (e2 < nnz) {
            int c2 = cols[e2];
            const float* s2 = (c2 < split) ? (srcA + (long)c2 * strA)
                                           : (srcB + (long)(c2 - split) * strB);
            __builtin_prefetch(s2 + lane * 4, 0, 0);
        }
        float4 x = *(const float4*)(s + lane * 4);
        float* d = dst + (long)r * strDst + lane * 4;
        atomicAdd(d + 0, v * x.x);
        atomicAdd(d + 1, v * x.y);
        atomicAdd(d + 2, v * x.z);
        atomicAdd(d + 3, v * x.w);
    }
}

// ---------------- gate MLP (WMMA) + gumbel-hard + emb combine ----------------
// concat = [linA | elu(nlB)]; MLP 256->64->32->16->2; emb = y0*linA + y1*elu(nlB)
__global__ void __launch_bounds__(256)
k_gate(const float* __restrict__ linA, int strA,
       const float* __restrict__ nlB, int strB,
       const _Float16* __restrict__ W1s, const _Float16* __restrict__ W2s,
       const _Float16* __restrict__ W3s, const _Float16* __restrict__ W4g,
       const float* __restrict__ b1, const float* __restrict__ b2,
       const float* __restrict__ b3, const float* __restrict__ b4,
       const float* __restrict__ noise, const float* __restrict__ temp_p,
       float* __restrict__ embOut, int strOut,
       float* __restrict__ cnts, int numTiles) {
    __shared__ _Float16 sH1[WAVES_PER_BLOCK][16 * 64];
    __shared__ _Float16 sH2[WAVES_PER_BLOCK][16 * 32];
    __shared__ float    sH3[WAVES_PER_BLOCK][16 * 16];
    __shared__ float    sY [WAVES_PER_BLOCK][32];

    int  tid     = threadIdx.x;
    int  lane    = tid & 31;
    int  wib     = tid >> 5;
    int  waveIdx = blockIdx.x * WAVES_PER_BLOCK + wib;
    bool active  = waveIdx < numTiles;
    int  tile    = active ? waveIdx : (numTiles - 1);   // clamp: EXEC all-1s, barriers uniform
    int  rowBase = tile * 16;
    int  r16     = lane & 15;
    int  kh      = (lane >> 4) * 8;
    float gtemp  = temp_p[0];

    // ---- layer 1: [16x256] @ [256x64], 8 K-chunks x 4 N-tiles of WMMA ----
    CAcc acc1[4];
#pragma unroll
    for (int t = 0; t < 4; ++t)
#pragma unroll
        for (int r = 0; r < 8; ++r) acc1[t].f[r] = 0.f;

    long rl = (long)(rowBase + r16);
    const float* pLin = linA + rl * strA;
    const float* pNl  = nlB + rl * strB;
#pragma unroll
    for (int c = 0; c < 8; ++c) {
        int kb = c * 32;
        const bool   isLin = (kb < 128);             // concat halves align to K-chunks
        const float* base  = isLin ? (pLin + kb) : (pNl + (kb - 128));
        _Float16 ah[16];
#pragma unroll
        for (int j = 0; j < 8; ++j) {                // two contiguous 8-elem runs
            float x0 = base[kh + j];
            float x1 = base[kh + 16 + j];
            if (!isLin) { x0 = elu_f(x0); x1 = elu_f(x1); }
            ah[j]     = (_Float16)x0;
            ah[8 + j] = (_Float16)x1;
        }
        v16h a = *(const v16h*)ah;
#pragma unroll
        for (int t = 0; t < 4; ++t) {
            v16h bfr = b_frag_swz(W1s, 4, c, t, lane);   // 2x global_load_b128, L2-hot
            acc1[t].v = __builtin_amdgcn_wmma_f32_16x16x32_f16(
                false, a, false, bfr, (short)0, acc1[t].v, false, false);
        }
    }
    _Float16* H1 = sH1[wib];
#pragma unroll
    for (int t = 0; t < 4; ++t) store_h_f16(H1, 64, lane, t * 16, acc1[t], b1);
    __syncthreads();

    // ---- layer 2: [16x64] @ [64x32] ----
    CAcc acc2[2];
#pragma unroll
    for (int t = 0; t < 2; ++t)
#pragma unroll
        for (int r = 0; r < 8; ++r) acc2[t].f[r] = 0.f;
#pragma unroll
    for (int c = 0; c < 2; ++c) {
        v16h a = a_frag_lds(H1, 64, lane, c * 32);
#pragma unroll
        for (int t = 0; t < 2; ++t) {
            v16h bfr = b_frag_swz(W2s, 2, c, t, lane);
            acc2[t].v = __builtin_amdgcn_wmma_f32_16x16x32_f16(
                false, a, false, bfr, (short)0, acc2[t].v, false, false);
        }
    }
    _Float16* H2 = sH2[wib];
#pragma unroll
    for (int t = 0; t < 2; ++t) store_h_f16(H2, 32, lane, t * 16, acc2[t], b2);
    __syncthreads();

    // ---- layer 3: [16x32] @ [32x16] ----
    CAcc acc3;
#pragma unroll
    for (int r = 0; r < 8; ++r) acc3.f[r] = 0.f;
    {
        v16h a   = a_frag_lds(H2, 32, lane, 0);
        v16h bfr = b_frag_swz(W3s, 1, 0, 0, lane);
        acc3.v = __builtin_amdgcn_wmma_f32_16x16x32_f16(
            false, a, false, bfr, (short)0, acc3.v, false, false);
    }
    store_h_f32(sH3[wib], 16, lane, 0, acc3, b3);
    __syncthreads();

    // ---- layer 4 (16->2) + gumbel softmax, hard argmax; lanes 0..15 = rows ----
    float* H3 = sH3[wib];
    float* Y  = sY[wib];
    bool pred = false;
    if (lane < 16) {
        float l0 = b4[0], l1 = b4[1];
#pragma unroll
        for (int k = 0; k < 16; ++k) {
            float h = H3[lane * 16 + k];
            l0 += h * (float)W4g[k * 2 + 0];
            l1 += h * (float)W4g[k * 2 + 1];
        }
        long  gr = (long)rowBase + lane;
        float z0 = (l0 + noise[gr * 2 + 0] * 0.125f) / gtemp;  // DIV_NOISE = 8
        float z1 = (l1 + noise[gr * 2 + 1] * 0.125f) / gtemp;
        float m  = fmaxf(z0, z1);
        float e0 = __expf(z0 - m), e1 = __expf(z1 - m);
        float y0 = e0 / (e0 + e1);
        pred = (y0 >= 1.f - y0);                 // argmax -> hard one-hot
        float y0h = pred ? 1.f : 0.f;
        Y[lane * 2 + 0] = y0h;
        Y[lane * 2 + 1] = 1.f - y0h;
    }
    // wave32 ballot reduction of hard selections -> 2 atomics per wave
    unsigned long long bal = __ballot(pred);
    if (lane == 0 && active) {
        int c0 = __popcll(bal & 0xFFFFull);
        atomicAdd(cnts + 0, (float)c0);
        atomicAdd(cnts + 1, (float)(16 - c0));
    }
    __syncthreads();

    // ---- emb = y0*lin + y1*elu(nl), 16 rows x 128 cols (float4 per lane) ----
    int cc = lane * 4;
    for (int r = 0; r < 16; ++r) {
        long  row = (long)rowBase + r;
        float y0 = Y[r * 2 + 0], y1 = Y[r * 2 + 1];
        float4 lv = *(const float4*)(linA + row * strA + cc);
        float4 nv = *(const float4*)(nlB + row * strB + cc);
        nv.x = elu_f(nv.x); nv.y = elu_f(nv.y); nv.z = elu_f(nv.z); nv.w = elu_f(nv.w);
        float4 o;
        o.x = y0 * lv.x + y1 * nv.x;
        o.y = y0 * lv.y + y1 * nv.y;
        o.z = y0 * lv.z + y1 * nv.z;
        o.w = y0 * lv.w + y1 * nv.w;
        if (active) *(float4*)(embOut + row * strOut + cc) = o;
    }
}

// ---------------- light_out = mean over 5 slots ----------------
__global__ void k_light(const float* __restrict__ embs, float* __restrict__ light) {
    long i = (long)blockIdx.x * blockDim.x + threadIdx.x;
    const long total = (long)NTOT * DIM;
    if (i < total) {
        int n = (int)(i / DIM), d = (int)(i % DIM);
        const float* row = embs + (long)n * EMB_STRIDE + d;
        light[i] = 0.2f * (row[0] + row[DIM] + row[2 * DIM] + row[3 * DIM] + row[4 * DIM]);
    }
}

// ---------------- gamma[b] = <light[u], light[NUSERS+it]> ----------------
__global__ void k_gamma(const float* __restrict__ light, const int* __restrict__ users,
                        const int* __restrict__ items, float* __restrict__ gamma) {
    int b = blockIdx.x * blockDim.x + threadIdx.x;
    if (b < BSZ) {
        const float* u = light + (long)users[b] * DIM;
        const float* v = light + (long)(NUSERS + items[b]) * DIM;
        float s = 0.f;
#pragma unroll 8
        for (int d = 0; d < DIM; d += 4) {
            float4 a = *(const float4*)(u + d);
            float4 c = *(const float4*)(v + d);
            s += a.x * c.x + a.y * c.y + a.z * c.z + a.w * c.w;
        }
        gamma[b] = s;
    }
}

// ---------------- driver ----------------
extern "C" void kernel_launch(void* const* d_in, const int* in_sizes, int n_in,
                              void* d_out, int out_size, void* d_ws, size_t ws_size,
                              hipStream_t stream) {
    (void)in_sizes; (void)n_in; (void)out_size; (void)ws_size;
    // jax-tree flattening order of setup_inputs():
    const float* user_emb = (const float*)d_in[0];
    const float* item_emb = (const float*)d_in[1];
    const float* g1W[4] = {(const float*)d_in[2], (const float*)d_in[4],
                           (const float*)d_in[6], (const float*)d_in[8]};
    const float* g1b[4] = {(const float*)d_in[3], (const float*)d_in[5],
                           (const float*)d_in[7], (const float*)d_in[9]};
    const float* g2W[4] = {(const float*)d_in[10], (const float*)d_in[12],
                           (const float*)d_in[14], (const float*)d_in[16]};
    const float* g2b[4] = {(const float*)d_in[11], (const float*)d_in[13],
                           (const float*)d_in[15], (const float*)d_in[17]};
    const float* noise1   = (const float*)d_in[18];
    const float* noise2   = (const float*)d_in[19];
    const float* evals    = (const float*)d_in[20];
    const int*   erow     = (const int*)d_in[21];
    const int*   ecol     = (const int*)d_in[22];
    const int*   users    = (const int*)d_in[23];
    const int*   items    = (const int*)d_in[24];
    const float* gum_temp = (const float*)d_in[25];

    float* out   = (float*)d_out;
    float* gamma = out;                 // [16384]
    float* cnts  = out + BSZ;           // lin3, non3, lin4, non4
    float* embs  = out + BSZ + 4;       // [N,5,128]

    float* wsF  = (float*)d_ws;
    float* lin3 = wsF;                        // N*128 (reused as light_out later)
    float* lin4 = wsF + (long)NTOT * DIM;     // N*128
    float* light = lin3;
    _Float16* wh = (_Float16*)(wsF + 2L * NTOT * DIM);  // 32B-aligned
    _Float16* g1h[4] = {wh, wh + 16384, wh + 16384 + 2048, wh + 16384 + 2048 + 512};
    _Float16* g2h[4] = {g1h[3] + 32, g1h[3] + 32 + 16384, g1h[3] + 32 + 16384 + 2048,
                        g1h[3] + 32 + 16384 + 2048 + 512};

    // convert + pre-swizzle weights (W1..W3 fragment-major; W4 plain f16)
    const int wsizes[3] = {256 * 64, 64 * 32, 32 * 16};
    const int wN[3]     = {64, 32, 16};
    for (int i = 0; i < 3; ++i) {
        k_swz_f16<<<(wsizes[i] + 255) / 256, 256, 0, stream>>>(g1W[i], g1h[i], wN[i], wsizes[i]);
        k_swz_f16<<<(wsizes[i] + 255) / 256, 256, 0, stream>>>(g2W[i], g2h[i], wN[i], wsizes[i]);
    }
    k_cvt_f16<<<1, 64, 0, stream>>>(g1W[3], g1h[3], 32);
    k_cvt_f16<<<1, 64, 0, stream>>>(g2W[3], g2h[3], 32);

    const long tot = (long)NTOT * DIM;
    k_init<<<(unsigned)((tot + 255) / 256), 256, 0, stream>>>(
        user_emb, item_emb, embs, lin3, lin4, cnts);

    const int SPMM_BLOCKS = 8192;
    const int IMAX = 0x7fffffff;
    // e_lin1 = spmm(e0)  (e0 split across user/item tables)
    k_spmm<<<SPMM_BLOCKS, 256, 0, stream>>>(user_emb, item_emb, NUSERS, DIM, DIM,
                                            evals, erow, ecol, embs + DIM, EMB_STRIDE, NNZ_K);
    // e_lin2 = spmm(e_lin1)
    k_spmm<<<SPMM_BLOCKS, 256, 0, stream>>>(embs + DIM, embs + DIM, IMAX, EMB_STRIDE, EMB_STRIDE,
                                            evals, erow, ecol, embs + 2 * DIM, EMB_STRIDE, NNZ_K);
    // e_lin3 = spmm(e_lin2) -> ws
    k_spmm<<<SPMM_BLOCKS, 256, 0, stream>>>(embs + 2 * DIM, embs + 2 * DIM, IMAX, EMB_STRIDE,
                                            EMB_STRIDE, evals, erow, ecol, lin3, DIM, NNZ_K);

    const int numTiles = NTOT / 16;  // 9375, exact
    const int gateBlocks = (numTiles + WAVES_PER_BLOCK - 1) / WAVES_PER_BLOCK;
    // gate1: concat=[e_lin3 | elu(e_lin1)] -> emb1 (slot 3)
    k_gate<<<gateBlocks, 256, 0, stream>>>(lin3, DIM, embs + DIM, EMB_STRIDE,
                                           g1h[0], g1h[1], g1h[2], g1h[3],
                                           g1b[0], g1b[1], g1b[2], g1b[3],
                                           noise1, gum_temp, embs + 3 * DIM, EMB_STRIDE,
                                           cnts + 0, numTiles);
    // e_lin4 = spmm(emb1) -> ws
    k_spmm<<<SPMM_BLOCKS, 256, 0, stream>>>(embs + 3 * DIM, embs + 3 * DIM, IMAX, EMB_STRIDE,
                                            EMB_STRIDE, evals, erow, ecol, lin4, DIM, NNZ_K);
    // gate2: concat=[e_lin4 | elu(e_lin4)] -> emb2 (slot 4)
    k_gate<<<gateBlocks, 256, 0, stream>>>(lin4, DIM, lin4, DIM,
                                           g2h[0], g2h[1], g2h[2], g2h[3],
                                           g2b[0], g2b[1], g2b[2], g2b[3],
                                           noise2, gum_temp, embs + 4 * DIM, EMB_STRIDE,
                                           cnts + 2, numTiles);

    k_light<<<(unsigned)((tot + 255) / 256), 256, 0, stream>>>(embs, light);
    k_gamma<<<(BSZ + 255) / 256, 256, 0, stream>>>(light, users, items, gamma);
}
